// ConcatRelationModule_88201448391415
// MI455X (gfx1250) — matrix operand order repaired and checked by reading, not compiled, not run on hardware
//
#include <hip/hip_runtime.h>

typedef __bf16 bf16;
typedef __attribute__((ext_vector_type(16))) __bf16 v16bf;
typedef __attribute__((ext_vector_type(8)))  __bf16 v8bf;
typedef __attribute__((ext_vector_type(8)))  float  v8f;

typedef int v4i __attribute__((vector_size(16)));
typedef __attribute__((address_space(1))) v4i* gptr_v4i;   // global AS pointer to int4
typedef __attribute__((address_space(3))) v4i* lptr_v4i;   // LDS AS pointer to int4

#define NTOK   16384
#define NEDGE  16383
#define LD     256
#define K1     512      // 2*LDIMS
#define N1     1024     // 2*HID
#define HID2   256
#define NREL   64

#define MTILE  64
#define TPB    128      // 4 waves of 32

// workspace offsets (bytes)
#define OFF_INB  0
#define SZ_INB   (NTOK*2*LD*2)
#define OFF_W1T  (OFF_INB + SZ_INB)
#define SZ_W1T   (N1*K1*2)
#define OFF_W2T  (OFF_W1T + SZ_W1T)
#define SZ_W2T   (HID2*N1*2)
#define OFF_W3T  (OFF_W2T + SZ_W2T)

// LDS layout (bf16 element pitches; pads keep 16B row alignment + bank skew)
#define PITCH_H  (N1 + 8)        // hs rows (64)           -> 2064 B/row
#define PITCH_S  (512 + 8)       // staging rows (64)      -> 1040 B/row
#define PITCH_H2 (HID2 + 8)      // h2s rows (64)          ->  528 B/row
#define SB_ELEMS     (64 * PITCH_S)                       // elems per staging buffer
#define SB_BYTES     (SB_ELEMS * 2)                       // 66560 B per buffer
#define LDS_HS_OFF   0
#define LDS_SB_OFF   (64 * PITCH_H * 2)                   // 132096
#define LDS_H2_OFF   (LDS_SB_OFF + 2 * SB_BYTES)          // 265216
#define LDS_BYTES    (LDS_H2_OFF + 64 * PITCH_H2 * 2)     // 299008  (< 320KB WGP)

#if __has_builtin(__builtin_amdgcn_global_load_async_to_lds_b128) && __has_builtin(__builtin_amdgcn_s_wait_asynccnt)
#define USE_ASYNC_LDS 1
#else
#define USE_ASYNC_LDS 0
#endif

union Frag { v16bf v; v8bf h[2]; };

template <int OFF>
__device__ __forceinline__ void async_cp(const bf16* g, bf16* l) {
#if USE_ASYNC_LDS
    // imm INST_OFFSET applies to both global and LDS address (ISA 15.18.3);
    // flat LDS pointer low 32 bits == LDS byte offset (aperture mapping)
    __builtin_amdgcn_global_load_async_to_lds_b128(
        (gptr_v4i)(unsigned long long)g,
        (lptr_v4i)(unsigned int)(unsigned long long)l,
        OFF, 0);
#else
    *(uint4*)((char*)l + OFF) = *(const uint4*)((const char*)g + OFF);
#endif
}

__device__ __forceinline__ void wait_staging() {
#if USE_ASYNC_LDS
    __builtin_amdgcn_s_wait_asynccnt(0);
#endif
}

// copy a 64-row x 512-col bf16 tile (row stride gstride elems) into an LDS buffer;
// each thread owns a contiguous 512B half-row -> 1 addr pair + imm offsets
__device__ __forceinline__ void stage_tile64(const bf16* __restrict__ g, int gstride,
                                             bf16* lbuf, int tid) {
    const int r  = tid >> 1;           // row 0..63
    const int ho = (tid & 1) * 256;    // elem offset of the 512B half-row
    const bf16* gb = g + (size_t)r * gstride + ho;
    bf16* lb = lbuf + r * PITCH_S + ho;
#define CP4(a) async_cp<(a)>(gb, lb); async_cp<(a)+16>(gb, lb); \
               async_cp<(a)+32>(gb, lb); async_cp<(a)+48>(gb, lb);
    CP4(0)   CP4(64)  CP4(128) CP4(192)
    CP4(256) CP4(320) CP4(384) CP4(448)
#undef CP4
}

__global__ void cvt_inputs_bf16(const float* __restrict__ src, bf16* __restrict__ dst, int n) {
    int i = blockIdx.x * blockDim.x + threadIdx.x;
    if (i < n) dst[i] = (bf16)src[i];
}

// src is [K][N] row-major f32; dst is [N][K] row-major bf16
__global__ void transpose_bf16(const float* __restrict__ src, bf16* __restrict__ dst, int K, int N) {
    int i = blockIdx.x * blockDim.x + threadIdx.x;
    if (i < K * N) {
        int k = i / N, n = i % N;
        dst[n * K + k] = (bf16)src[i];
    }
}

__global__ __launch_bounds__(TPB) void relmlp_main(
    const bf16* __restrict__ inb,     // [NTOK][2][LD] bf16
    const bf16* __restrict__ w1t,     // [N1][K1]
    const bf16* __restrict__ w2t,     // [HID2][N1]
    const bf16* __restrict__ w3t,     // [NREL][HID2]
    const float* __restrict__ bcat,   // [N1]
    const float* __restrict__ b2,     // [HID2]
    const float* __restrict__ b3,     // [NREL]
    const int* __restrict__ heads,    // [NEDGE]
    float* __restrict__ out)          // [NEDGE][NREL]
{
    extern __shared__ char smem[];
    bf16* hs  = (bf16*)(smem + LDS_HS_OFF);   // [64][PITCH_H]
    bf16* sb  = (bf16*)(smem + LDS_SB_OFF);   // [2][64][PITCH_S] double buffer
    bf16* h2s = (bf16*)(smem + LDS_H2_OFF);   // [64][PITCH_H2]

    const int tid  = threadIdx.x;
    const int wave = tid >> 5;
    const int lane = tid & 31;
    const int lrow = lane & 15;   // fragment row (A) / col (B,C)
    const int lhi  = lane >> 4;   // half-wave select

    const int ebase = blockIdx.x * MTILE;
    const int rowW  = wave * 16;  // wave's row base inside tile

    // ---- gather + preload layer-1 A fragments (this lane's edge row) ----
    int e = ebase + rowW + lrow;
    if (e >= NEDGE) e = NEDGE - 1;
    const int hidx = heads[e];
    const bf16* phead = inb + ((size_t)hidx * 2 + 0) * LD;    // K 0..255
    const bf16* pmod  = inb + ((size_t)(e + 1) * 2 + 1) * LD; // K 256..511

    Frag xf[16];
    #pragma unroll
    for (int c = 0; c < 16; ++c) {
        int kb = c * 32;
        const bf16* src = (kb < LD) ? (phead + kb) : (pmod + (kb - LD));
        const int lo = lhi * 8;
        xf[c].h[0] = *(const v8bf*)(src + lo);        // K lo..lo+7
        xf[c].h[1] = *(const v8bf*)(src + lo + 16);   // K lo+16..lo+23
    }

    // =============== Layer 1: h = tanh(x @ W1 + bcat), N1=1024 ===============
    // quads of N-tiles: 16 iterations, staged tile = rows [64nq,64nq+64) x K1
    stage_tile64(w1t, K1, sb, tid);
    #pragma unroll 1
    for (int nq = 0; nq < N1 / 64; ++nq) {
        wait_staging();
        __syncthreads();
        if (nq + 1 < N1 / 64)
            stage_tile64(w1t + (size_t)(nq + 1) * 64 * K1, K1,
                         sb + ((nq + 1) & 1) * SB_ELEMS, tid);
        bf16* buf = sb + (nq & 1) * SB_ELEMS;   // integer offset keeps AS3 inference

        v8f acc[4];
        #pragma unroll
        for (int q = 0; q < 4; ++q) {
            const float b = bcat[nq * 64 + q * 16 + lrow];
            #pragma unroll
            for (int i = 0; i < 8; ++i) acc[q][i] = b;
        }

        #pragma unroll
        for (int c = 0; c < 16; ++c) {
            const int kb = c * 32;
            Frag bfr[4];
            #pragma unroll
            for (int q = 0; q < 4; ++q) {
                const bf16* br = buf + (q * 16 + lrow) * PITCH_S + lhi * 8 + kb;
                bfr[q].h[0] = *(const v8bf*)(br);
                bfr[q].h[1] = *(const v8bf*)(br + 16);
            }
            #pragma unroll
            for (int q = 0; q < 4; ++q)
                acc[q] = __builtin_amdgcn_wmma_f32_16x16x32_bf16(
                    false, xf[c].v, false, bfr[q].v, (short)0, acc[q], false, false);
        }
        const int mrow = rowW + lhi * 8;
        #pragma unroll
        for (int q = 0; q < 4; ++q)
            #pragma unroll
            for (int v = 0; v < 8; ++v)
                hs[(mrow + v) * PITCH_H + nq * 64 + q * 16 + lrow] = (bf16)tanhf(acc[q][v]);
    }

    // =============== Layer 2: h2 = tanh(h @ W2 + b2), HID2=256 ===============
    // 4 N-quads x 2 K-halves: staged tile = rows [64nq,64nq+64) x cols [512kh,512kh+512)
    stage_tile64(w2t, N1, sb, tid);
    v8f acc[4];
    #pragma unroll 1
    for (int s = 0; s < 8; ++s) {
        const int nq = s >> 1, kh = s & 1;
        if (kh == 0) {
            #pragma unroll
            for (int q = 0; q < 4; ++q) {
                const float b = b2[nq * 64 + q * 16 + lrow];
                #pragma unroll
                for (int i = 0; i < 8; ++i) acc[q][i] = b;
            }
        }
        wait_staging();
        __syncthreads();
        if (s + 1 < 8) {
            const int nq2 = (s + 1) >> 1, kh2 = (s + 1) & 1;
            stage_tile64(w2t + (size_t)nq2 * 64 * N1 + kh2 * 512, N1,
                         sb + ((s + 1) & 1) * SB_ELEMS, tid);
        }
        bf16* buf = sb + (s & 1) * SB_ELEMS;

        const bf16* arow = hs + (rowW + lrow) * PITCH_H + kh * 512;
        #pragma unroll
        for (int c = 0; c < 16; ++c) {
            const int kb = c * 32;
            Frag afr, bfr[4];
            afr.h[0] = *(const v8bf*)(arow + kb + lhi * 8);
            afr.h[1] = *(const v8bf*)(arow + kb + lhi * 8 + 16);
            #pragma unroll
            for (int q = 0; q < 4; ++q) {
                const bf16* br = buf + (q * 16 + lrow) * PITCH_S + lhi * 8 + kb;
                bfr[q].h[0] = *(const v8bf*)(br);
                bfr[q].h[1] = *(const v8bf*)(br + 16);
            }
            #pragma unroll
            for (int q = 0; q < 4; ++q)
                acc[q] = __builtin_amdgcn_wmma_f32_16x16x32_bf16(
                    false, afr.v, false, bfr[q].v, (short)0, acc[q], false, false);
        }
        if (kh == 1) {
            const int mrow = rowW + lhi * 8;
            #pragma unroll
            for (int q = 0; q < 4; ++q)
                #pragma unroll
                for (int v = 0; v < 8; ++v)
                    h2s[(mrow + v) * PITCH_H2 + nq * 64 + q * 16 + lrow] = (bf16)tanhf(acc[q][v]);
        }
    }

    // =============== Layer 3: out = h2 @ W3 + b3, NREL=64 (one quad) ===============
    {
        v8f c4[4];
        #pragma unroll
        for (int q = 0; q < 4; ++q) {
            const float b = b3[q * 16 + lrow];
            #pragma unroll
            for (int i = 0; i < 8; ++i) c4[q][i] = b;
        }
        const bf16* arow = h2s + (rowW + lrow) * PITCH_H2;
        #pragma unroll
        for (int c = 0; c < 8; ++c) {
            const int kb = c * 32;
            Frag afr, bfr[4];
            afr.h[0] = *(const v8bf*)(arow + kb + lhi * 8);
            afr.h[1] = *(const v8bf*)(arow + kb + lhi * 8 + 16);
            #pragma unroll
            for (int q = 0; q < 4; ++q) {
                const bf16* br = w3t + (size_t)(q * 16 + lrow) * HID2 + lhi * 8 + kb; // L2-hot
                bfr[q].h[0] = *(const v8bf*)(br);
                bfr[q].h[1] = *(const v8bf*)(br + 16);
            }
            #pragma unroll
            for (int q = 0; q < 4; ++q)
                c4[q] = __builtin_amdgcn_wmma_f32_16x16x32_bf16(
                    false, afr.v, false, bfr[q].v, (short)0, c4[q], false, false);
        }
        const int mrow = rowW + lhi * 8;
        #pragma unroll
        for (int v = 0; v < 8; ++v) {
            const int ee = ebase + mrow + v;
            if (ee < NEDGE) {
                #pragma unroll
                for (int q = 0; q < 4; ++q)
                    out[(size_t)ee * NREL + q * 16 + lrow] = c4[q][v];
            }
        }
    }
}

extern "C" void kernel_launch(void* const* d_in, const int* in_sizes, int n_in,
                              void* d_out, int out_size, void* d_ws, size_t ws_size,
                              hipStream_t stream) {
    const float* inputs = (const float*)d_in[0];
    const float* foh    = (const float*)d_in[1];
    const float* fom    = (const float*)d_in[2];
    const float* bcat   = (const float*)d_in[3];
    const float* w2     = (const float*)d_in[4];
    const float* b2     = (const float*)d_in[5];
    const float* w3     = (const float*)d_in[6];
    const float* b3     = (const float*)d_in[7];
    const int*   heads  = (const int*)d_in[8];

    char* ws = (char*)d_ws;
    bf16* inb = (bf16*)(ws + OFF_INB);
    bf16* w1t = (bf16*)(ws + OFF_W1T);
    bf16* w2t = (bf16*)(ws + OFF_W2T);
    bf16* w3t = (bf16*)(ws + OFF_W3T);

    // prep: f32 -> bf16 conversions (weights transposed to N-major)
    const int nin = NTOK * 2 * LD;
    cvt_inputs_bf16<<<(nin + 255) / 256, 256, 0, stream>>>(inputs, inb, nin);
    transpose_bf16<<<(K1 * 512 + 255) / 256, 256, 0, stream>>>(foh, w1t, K1, 512);
    transpose_bf16<<<(K1 * 512 + 255) / 256, 256, 0, stream>>>(fom, w1t + 512 * K1, K1, 512);
    transpose_bf16<<<(N1 * HID2 + 255) / 256, 256, 0, stream>>>(w2, w2t, N1, HID2);
    transpose_bf16<<<(HID2 * NREL + 255) / 256, 256, 0, stream>>>(w3, w3t, HID2, NREL);

    const int nblocks = (NEDGE + MTILE - 1) / MTILE;  // 256
    relmlp_main<<<nblocks, TPB, LDS_BYTES, stream>>>(
        inb, w1t, w2t, w3t, bcat, b2, b3, heads, (float*)d_out);
}
